// GraphAttentionLayer_39694087749738
// MI455X (gfx1250) — matrix operand order repaired and checked by reading, not compiled, not run on hardware
//
#include <hip/hip_runtime.h>

// GAT layer fused kernel for gfx1250 (MI455X), wave32 + v_wmma_f32_16x16x32_bf16.
// Phase 1: Wh = h@W (f32 FMA, tiny); s1/s2 row scores stored pre-scaled by
//          log2(e); WhT stored transposed in bf16 for contiguous B-fragments.
// Phase 1b: s2max reduction (upper bound for a FIXED per-row softmax shift).
// Phase 2: fused masked-softmax(adj) @ Wh. Fixed shift M[m]=leakyrelu(s1+s2max)
//          => no online rescale; 4-way column split merges by addition in LDS.
//          Hot loop unrolled x2 with ping-pong buffers: loads for one chunk in
//          flight while the other computes; no rotation copies.

#define NROWS   8192
#define IN_F    128
#define OUT_F   64
#define ALPHA_F 0.2f
#define LOG2E_F 1.4426950408889634f
#define CSTEP   128

typedef __attribute__((ext_vector_type(16))) __bf16 v16bf;
typedef __attribute__((ext_vector_type(8)))  float  v8f;

// ---------------------------------------------------------------------------
// Phase 1: one wave per row. Lane f computes Wh[row][f] and Wh[row][f+32],
// wave-reduces s1/s2 (stored *LOG2E), stores WhT[feat][row] (bf16).
// ---------------------------------------------------------------------------
__global__ __launch_bounds__(256) void gat_precompute(
    const float* __restrict__ h, const float* __restrict__ W,
    const float* __restrict__ a, __bf16* __restrict__ whT,
    float* __restrict__ s1, float* __restrict__ s2) {
  const int lane = threadIdx.x & 31;
  const int row  = blockIdx.x * (blockDim.x >> 5) + (threadIdx.x >> 5);
  if (row >= NROWS) return;

  const float* hr = h + (size_t)row * IN_F;
  float w0 = 0.f, w1 = 0.f;
#pragma unroll 4
  for (int k = 0; k < IN_F; ++k) {
    const float hv = hr[k];
    w0 = fmaf(hv, W[k * OUT_F + lane], w0);
    w1 = fmaf(hv, W[k * OUT_F + lane + 32], w1);
  }
  float p1 = w0 * a[lane]      + w1 * a[lane + 32];
  float p2 = w0 * a[64 + lane] + w1 * a[96 + lane];
#pragma unroll
  for (int off = 16; off > 0; off >>= 1) {
    p1 += __shfl_xor(p1, off, 32);
    p2 += __shfl_xor(p2, off, 32);
  }
  if (lane == 0) { s1[row] = p1 * LOG2E_F; s2[row] = p2 * LOG2E_F; }
  whT[(size_t)lane        * NROWS + row] = (__bf16)w0;
  whT[(size_t)(lane + 32) * NROWS + row] = (__bf16)w1;
}

// ---------------------------------------------------------------------------
// Phase 1b: max over s2 (scaled domain), single block.
// ---------------------------------------------------------------------------
__global__ __launch_bounds__(256) void gat_s2max(
    const float* __restrict__ s2, float* __restrict__ s2max) {
  __shared__ float red[256];
  float mx = -__builtin_inff();
  for (int i = threadIdx.x; i < NROWS; i += 256) mx = fmaxf(mx, s2[i]);
  red[threadIdx.x] = mx;
  __syncthreads();
#pragma unroll
  for (int s = 128; s > 0; s >>= 1) {
    if (threadIdx.x < s) red[threadIdx.x] = fmaxf(red[threadIdx.x], red[threadIdx.x + s]);
    __syncthreads();
  }
  if (threadIdx.x == 0) *s2max = red[0];
}

// ---------------------------------------------------------------------------
// Phase 2 helpers: one 16x32 chunk = adj mask (4x int4), s2' (4x float4),
// 4 B-fragments (16 contiguous bf16 per lane from WhT).
// ---------------------------------------------------------------------------
struct Chunk {
  int   av[16];
  float sv[16];
  v16bf Bt[4];
};

// ---------------------------------------------------------------------------
// Phase 2: 4 waves/block share one 16-row block, splitting the 8192 columns
// 4-way (wave w takes chunks c = w*32, step 128). Per 16x32 chunk:
//   p = exp2( max(cA + s2', fma(0.2, s2', cF)) ) masked by adj  (log2 domain)
//   A (16x32 bf16) packed in-register -> 4x v_wmma_f32_16x16x32_bf16.
// ---------------------------------------------------------------------------
__global__ __launch_bounds__(128) void gat_attention(
    const int* __restrict__ adj, const __bf16* __restrict__ whT,
    const float* __restrict__ s1, const float* __restrict__ s2,
    const float* __restrict__ s2max, float* __restrict__ out) {
  __shared__ float lsAcc[4][32][32];  // [wave][lane][tile*8+j]
  __shared__ float lsL[4][32];

  const int lane = threadIdx.x & 31;
  const int wave = threadIdx.x >> 5;
  const int row0 = blockIdx.x * 16;
  const int m    = lane & 15;          // row within block / feature lane
  const int hi   = lane >> 4;          // K-half select for A/B layout
  const int row  = row0 + m;

  // Fixed per-row shift: M = leakyrelu(s1+s2max) >= all unmasked scores (log2 dom).
  const float s1m = s1[row];
  const float u   = s1m + *s2max;
  const float Mm  = fmaxf(u, ALPHA_F * u);
  const float cA  = s1m - Mm;              // pos branch:  cA + s2'
  const float cF  = ALPHA_F * s1m - Mm;    // neg branch:  fma(0.2, s2', cF)

  const int*    adjr = adj + (size_t)row * NROWS;
  const __bf16* wbB  = whT + (size_t)m * NROWS + hi * 16;

  v8f acc[4];
#pragma unroll
  for (int t = 0; t < 4; ++t) acc[t] = (v8f){0.f,0.f,0.f,0.f,0.f,0.f,0.f,0.f};
  float psa[4] = {0.f, 0.f, 0.f, 0.f};     // break the row-sum dep chain

  auto loadChunk = [&](int c, Chunk& b) __attribute__((always_inline)) {
    const int cb = c + hi * 16;
    __builtin_prefetch(adjr + cb + 2048, 0, 0);   // global_prefetch_b8, adj stream
    *(int4*)(b.av + 0)  = *(const int4*)(adjr + cb);
    *(int4*)(b.av + 4)  = *(const int4*)(adjr + cb + 4);
    *(int4*)(b.av + 8)  = *(const int4*)(adjr + cb + 8);
    *(int4*)(b.av + 12) = *(const int4*)(adjr + cb + 12);
    *(float4*)(b.sv + 0)  = *(const float4*)(s2 + cb);
    *(float4*)(b.sv + 4)  = *(const float4*)(s2 + cb + 4);
    *(float4*)(b.sv + 8)  = *(const float4*)(s2 + cb + 8);
    *(float4*)(b.sv + 12) = *(const float4*)(s2 + cb + 12);
    const __bf16* wb = wbB + c;
#pragma unroll
    for (int t = 0; t < 4; ++t)
      b.Bt[t] = *(const v16bf*)(wb + (size_t)(16 * t) * NROWS);
  };

  auto computeChunk = [&](const Chunk& b) __attribute__((always_inline)) {
    v16bf A;
#pragma unroll
    for (int i = 0; i < 16; ++i) {
      const float t0 = cA + b.sv[i];                   // v_add
      const float t1 = fmaf(ALPHA_F, b.sv[i], cF);     // v_fma
      float p = __builtin_amdgcn_exp2f(fmaxf(t0, t1)); // v_max + v_exp (2^x)
      p = (b.av[i] > 0) ? p : 0.f;                     // mask
      psa[i & 3] += p;
      A[i] = (__bf16)p;
    }
#pragma unroll
    for (int t = 0; t < 4; ++t)
      acc[t] = __builtin_amdgcn_wmma_f32_16x16x32_bf16(
          false, A, false, b.Bt[t], (short)0, acc[t], false, false);
  };

  const int cstart = wave * 32;
  Chunk b0, b1;
  loadChunk(cstart, b0);

  // 64 chunks per wave, unrolled x2 ping-pong. With cstart <= 96 the c+CSTEP
  // chunk is always < NROWS; only the c+2*CSTEP reload needs the (uniform)
  // wrap select, which keeps the tail load in-bounds and EXEC full.
  for (int c = cstart; c < NROWS; c += 2 * CSTEP) {
    loadChunk(c + CSTEP, b1);
    computeChunk(b0);
    const int c2 = (c + 2 * CSTEP < NROWS) ? (c + 2 * CSTEP) : cstart;
    loadChunk(c2, b0);
    computeChunk(b1);
  }

  float ps = (psa[0] + psa[1]) + (psa[2] + psa[3]);
  ps += __shfl_xor(ps, 16, 32);   // combine the two K-halves of each row

  // ---- merge the 4 column splits by addition via LDS ----
  lsL[wave][lane] = ps;
#pragma unroll
  for (int t = 0; t < 4; ++t)
#pragma unroll
    for (int j = 0; j < 8; ++j) lsAcc[wave][lane][t * 8 + j] = acc[t][j];
  __syncthreads();

  if (wave == 0) {
    float Lt = ps;
#pragma unroll
    for (int w = 1; w < 4; ++w) {
      Lt += lsL[w][lane];
#pragma unroll
      for (int t = 0; t < 4; ++t)
#pragma unroll
        for (int j = 0; j < 8; ++j) acc[t][j] += lsAcc[w][lane][t * 8 + j];
    }

    float Lr[8];
#pragma unroll
    for (int j = 0; j < 8; ++j) Lr[j] = __shfl(Lt, j + (hi << 3), 32);

#pragma unroll
    for (int t = 0; t < 4; ++t) {
#pragma unroll
      for (int j = 0; j < 8; ++j) {
        const float hp = acc[t][j] / Lr[j];
        const float o  = hp > 0.f ? hp : (__expf(hp) - 1.f);  // elu
        const int orow = row0 + j + (hi << 3);
        out[(size_t)orow * OUT_F + 16 * t + m] = o;
      }
    }
  }
}

// ---------------------------------------------------------------------------
extern "C" void kernel_launch(void* const* d_in, const int* in_sizes, int n_in,
                              void* d_out, int out_size, void* d_ws, size_t ws_size,
                              hipStream_t stream) {
  const float* h   = (const float*)d_in[0];
  const int*   adj = (const int*)d_in[1];
  const float* W   = (const float*)d_in[2];
  const float* a   = (const float*)d_in[3];
  float*       out = (float*)d_out;

  char*   ws    = (char*)d_ws;
  __bf16* whT   = (__bf16*)ws;                                  // 1 MB
  float*  s1    = (float*)(ws + (size_t)OUT_F * NROWS * 2);     // 32 KB
  float*  s2    = s1 + NROWS;                                   // 32 KB
  float*  s2max = s2 + NROWS;                                   // 4 B

  gat_precompute<<<NROWS / 8, 256, 0, stream>>>(h, W, a, whT, s1, s2);
  gat_s2max<<<1, 256, 0, stream>>>(s2, s2max);
  // 512 row-blocks x 4 column splits = 2048 waves.
  gat_attention<<<NROWS / 16, 128, 0, stream>>>(adj, whT, s1, s2, s2max, out);
}
